// SparseMultiheadAttention_12317966205231
// MI455X (gfx1250) — matrix operand order repaired and checked by reading, not compiled, not run on hardware
//
#include <hip/hip_runtime.h>
#include <hip/hip_bf16.h>

// Problem dimensions (fixed by the reference)
static constexpr int Bc = 4, Tc = 2048, Sc = 2048, Ec = 512, Hc = 8, Dc = 64;
static constexpr int KEEP = Sc / 2;   // SPARSITY = 0.5

typedef __attribute__((ext_vector_type(16))) __bf16 v16bf;
typedef __attribute__((ext_vector_type(8)))  float  v8f;

union Frag16 {            // 32-byte WMMA A/B fragment, assembled as dwords
  v16bf v;
  unsigned u[8];
  uint4 q[2];
};

__device__ __forceinline__ __bf16 f2bf(float f) {
  unsigned u = __float_as_uint(f);
  unsigned r = u + 0x7FFFu + ((u >> 16) & 1u);           // round-to-nearest-even
  unsigned short hs = (unsigned short)(r >> 16);
  return __builtin_bit_cast(__bf16, hs);
}

// pack two fp32 -> one dword of two bf16 via the HW packed convert (1 VALU op)
__device__ __forceinline__ unsigned pack_bf2(float a, float b) {
  unsigned r;
  asm("v_cvt_pk_bf16_f32 %0, %1, %2" : "=v"(r) : "v"(a), "v"(b));
  return r;
}

__device__ __forceinline__ v8f wmma_bf16(v16bf a, v16bf b, v8f c) {
  return __builtin_amdgcn_wmma_f32_16x16x32_bf16(false, a, false, b, (short)0, c,
                                                 false, false);
}

// ---- fragment loaders (ISA 7.12.2 layouts, wave32) -------------------------
// A: 16x32 bf16; lane m = lane&15, K chunks {kb..kb+7, kb+16..kb+23}, kb=(lane>>4)*8
__device__ __forceinline__ v16bf loadA_f32(const float* base, int ld, int k0) {
  int lane = threadIdx.x & 31;
  const float* p = base + (size_t)(lane & 15) * ld + k0 + ((lane >> 4) * 8);
  Frag16 f;
#pragma unroll
  for (int j = 0; j < 4; ++j) f.u[j]     = pack_bf2(p[2 * j],      p[2 * j + 1]);
#pragma unroll
  for (int j = 0; j < 4; ++j) f.u[4 + j] = pack_bf2(p[16 + 2 * j], p[16 + 2 * j + 1]);
  return f.v;
}

__device__ __forceinline__ v16bf loadA_bf(const __bf16* base, int ld, int k0) {
  int lane = threadIdx.x & 31;
  const __bf16* p = base + (size_t)(lane & 15) * ld + k0 + ((lane >> 4) * 8);
  Frag16 f;
  f.q[0] = *(const uint4*)(p);
  f.q[1] = *(const uint4*)(p + 16);
  return f.v;
}

// B: 32x16 bf16 from Bt row-major [n][k]; lane n = lane&15, K chunks {kb..kb+7, kb+16..kb+23}, kb=(lane>>4)*8
__device__ __forceinline__ v16bf loadB_f32(const float* base, int ld, int k0) {
  int lane = threadIdx.x & 31;
  const float* p = base + (size_t)(lane & 15) * ld + k0 + ((lane >> 4) * 8);
  Frag16 f;
#pragma unroll
  for (int j = 0; j < 4; ++j) f.u[j]     = pack_bf2(p[2 * j],      p[2 * j + 1]);
#pragma unroll
  for (int j = 0; j < 4; ++j) f.u[4 + j] = pack_bf2(p[16 + 2 * j], p[16 + 2 * j + 1]);
  return f.v;
}

__device__ __forceinline__ v16bf loadB_bf(const __bf16* base, int ld, int k0) {
  int lane = threadIdx.x & 31;
  const __bf16* p = base + (size_t)(lane & 15) * ld + k0 + ((lane >> 4) * 8);
  Frag16 f;
  f.q[0] = *(const uint4*)(p);
  f.q[1] = *(const uint4*)(p + 16);
  return f.v;
}

// ---- K1: y = x @ W^T (fp32 in, bf16 out), scatter to head layouts ----------
// mode 0/1: out (B,H,T,D) row-major D   (Q and K)
// mode 2  : out (B,H,D,S) row-major S   (V transposed)
__global__ __launch_bounds__(256)
void qkv_proj_kernel(const float* __restrict__ x, const float* __restrict__ w,
                     __bf16* __restrict__ out, int mode) {
  int n_blk = blockIdx.x & 3;
  int m_blk = blockIdx.x >> 2;
  int wave = threadIdx.x >> 5;
  int m0 = m_blk * 16, n0 = n_blk * 128 + wave * 16;
  const float* pa = x + (size_t)m0 * Ec;
  const float* pb = w + (size_t)n0 * Ec;
  v8f c = {0.f, 0.f, 0.f, 0.f, 0.f, 0.f, 0.f, 0.f};
  for (int k0 = 0; k0 < Ec; k0 += 32) {
    __builtin_prefetch(pa + k0 + 64, 0, 0);
    c = wmma_bf16(loadA_f32(pa, Ec, k0), loadB_f32(pb, Ec, k0), c);
  }
  int lane = threadIdx.x & 31;
  int hi = lane >> 4, nn = lane & 15;
#pragma unroll
  for (int r = 0; r < 8; ++r) {
    int rowg = m0 + hi * 8 + r;          // row in B*T (T==S)
    int b = rowg / Tc, t = rowg % Tc;
    int col = n0 + nn;
    int h = col / Dc, dd = col % Dc;
    __bf16 v = f2bf(c[r]);
    if (mode == 2)
      out[(((size_t)(b * Hc + h)) * Dc + dd) * Sc + t] = v;
    else
      out[(((size_t)(b * Hc + h)) * Tc + t) * Dc + dd] = v;
  }
}

// ---- K2: scores = (Q K^T) * 1/sqrt(D) -> fp32 into d_out weights region ----
// Q tile (16x64 bf16, shared by all 8 waves) staged once via async load to LDS.
__global__ __launch_bounds__(256)
void scores_kernel(const __bf16* __restrict__ q, const __bf16* __restrict__ k,
                   float* __restrict__ wts) {
  __shared__ alignas(16) __bf16 sq[16 * Dc];   // 2 KB
  int bid = blockIdx.x;
  int s_blk = bid & 15; bid >>= 4;
  int t_blk = bid & 127; int bh = bid >> 7;
  int wave = threadIdx.x >> 5;
  int tid = threadIdx.x;
  int t0 = t_blk * 16, s0 = s_blk * 128 + wave * 16;

  // async-stage Q tile: 256 lanes x 8B = 2KB, direct to LDS (bypasses VGPRs)
  {
    const __bf16* g = q + ((size_t)bh * Tc + t0) * Dc + tid * 4;
    unsigned lds = (unsigned)(uintptr_t)(&sq[tid * 4]);
    asm volatile("global_load_async_to_lds_b64 %0, %1, off"
                 :: "v"(lds), "v"(g) : "memory");
    asm volatile("s_wait_asynccnt 0x0" ::: "memory");
  }
  __syncthreads();

  const __bf16* pb = k + ((size_t)bh * Sc + s0) * Dc;
  v8f c = {0.f, 0.f, 0.f, 0.f, 0.f, 0.f, 0.f, 0.f};
#pragma unroll
  for (int k0 = 0; k0 < Dc; k0 += 32)
    c = wmma_bf16(loadA_bf(sq, Dc, k0), loadB_bf(pb, Dc, k0), c);

  int lane = threadIdx.x & 31;
  int hi = lane >> 4, n = lane & 15;
  float* o = wts + ((size_t)bh * Tc + t0) * Sc + s0 + n;
#pragma unroll
  for (int r = 0; r < 8; ++r)
    o[(size_t)(hi * 8 + r) * Sc] = c[r] * 0.125f;   // 1/sqrt(64)
}

// ---- K3: per-row exact top-k (radix select) + masked softmax, in place -----
__global__ __launch_bounds__(256)
void topk_softmax_kernel(float* __restrict__ wts) {
  __shared__ unsigned su[Sc];
  __shared__ float sf[Sc];
  __shared__ float sred[256];
  __shared__ int scnt;
  int tid = threadIdx.x;
  float* row = wts + (size_t)blockIdx.x * Sc;
  for (int i = tid; i < Sc; i += 256) {
    float f = row[i];
    unsigned b = __float_as_uint(f);
    su[i] = (b & 0x80000000u) ? ~b : (b | 0x80000000u);  // monotonic key
    sf[i] = f;
  }
  __syncthreads();
  unsigned prefix = 0;
  int kk = KEEP;
  for (int bit = 31; bit >= 0; --bit) {
    if (tid == 0) scnt = 0;
    __syncthreads();
    unsigned target = (prefix >> bit) | 1u;
    int local = 0;
    for (int i = tid; i < Sc; i += 256) local += ((su[i] >> bit) == target);
    atomicAdd(&scnt, local);
    __syncthreads();
    int c1 = scnt;
    if (c1 >= kk) prefix |= (1u << bit); else kk -= c1;
    __syncthreads();
  }
  unsigned uth = prefix;                 // key of the KEEP-th largest value
  float m = -3.4e38f;
  for (int i = tid; i < Sc; i += 256)
    if (su[i] >= uth) m = fmaxf(m, sf[i]);
  sred[tid] = m; __syncthreads();
  for (int s = 128; s > 0; s >>= 1) {
    if (tid < s) sred[tid] = fmaxf(sred[tid], sred[tid + s]);
    __syncthreads();
  }
  m = sred[0]; __syncthreads();
  float acc = 0.f;
  for (int i = tid; i < Sc; i += 256)
    if (su[i] >= uth) acc += __expf(sf[i] - m);
  sred[tid] = acc; __syncthreads();
  for (int s = 128; s > 0; s >>= 1) {
    if (tid < s) sred[tid] += sred[tid + s];
    __syncthreads();
  }
  float inv = 1.0f / sred[0];
  __syncthreads();
  for (int i = tid; i < Sc; i += 256)
    row[i] = (su[i] >= uth) ? __expf(sf[i] - m) * inv : 0.0f;
}

// ---- K4: context = P @ V ---------------------------------------------------
// The 16x32 fp32 P tile is shared by all 4 waves each K-step: double-buffered
// async stage to LDS (global_load_async_to_lds_b128), overlapped with WMMA.
__global__ __launch_bounds__(128)
void context_kernel(const float* __restrict__ wts, const __bf16* __restrict__ vt,
                    __bf16* __restrict__ ctx) {
  __shared__ alignas(16) float sa[2][16 * 32];  // 2 x 2 KB
  int bid = blockIdx.x;
  int t_blk = bid & 127; int bh = bid >> 7;
  int b = bh >> 3, h = bh & 7;
  int wave = threadIdx.x >> 5;
  int tid = threadIdx.x;
  int t0 = t_blk * 16, d0 = wave * 16;
  const float* pa = wts + ((size_t)bh * Tc + t0) * Sc;
  const __bf16* pb = vt + ((size_t)bh * Dc + d0) * Sc;

  int arow = tid >> 3, achk = (tid & 7) * 4;    // 16 rows x 8 chunks of 16B

  // issue tile for s0 into buffer `buf`
  auto issue = [&](int buf, int s0) {
    const float* g = pa + (size_t)arow * Sc + s0 + achk;
    unsigned lds = (unsigned)(uintptr_t)(&sa[buf][arow * 32 + achk]);
    asm volatile("global_load_async_to_lds_b128 %0, %1, off"
                 :: "v"(lds), "v"(g) : "memory");
  };

  issue(0, 0);
  int cur = 0;
  v8f c = {0.f, 0.f, 0.f, 0.f, 0.f, 0.f, 0.f, 0.f};
  for (int s0 = 0; s0 < Sc; s0 += 32) {
    if (s0 + 32 < Sc) {
      issue(cur ^ 1, s0 + 32);
      asm volatile("s_wait_asynccnt 0x1" ::: "memory");
    } else {
      asm volatile("s_wait_asynccnt 0x0" ::: "memory");
    }
    __syncthreads();
    c = wmma_bf16(loadA_f32(&sa[cur][0], 32, 0), loadB_bf(pb, Sc, s0), c);
    __syncthreads();
    cur ^= 1;
  }

  int lane = threadIdx.x & 31;
  int hi = lane >> 4, n = lane & 15;
#pragma unroll
  for (int r = 0; r < 8; ++r) {
    int t = t0 + hi * 8 + r;
    int col = h * Dc + d0 + n;
    ctx[((size_t)(b * Tc + t)) * Ec + col] = f2bf(c[r]);
  }
}

// ---- K5: attn_out = ctx @ w_o^T -> fp32 d_out ------------------------------
__global__ __launch_bounds__(256)
void oproj_kernel(const __bf16* __restrict__ ctx, const float* __restrict__ wo,
                  float* __restrict__ out) {
  int n_blk = blockIdx.x & 3;
  int m_blk = blockIdx.x >> 2;
  int wave = threadIdx.x >> 5;
  int m0 = m_blk * 16, n0 = n_blk * 128 + wave * 16;
  const __bf16* pa = ctx + (size_t)m0 * Ec;
  const float* pb = wo + (size_t)n0 * Ec;
  v8f c = {0.f, 0.f, 0.f, 0.f, 0.f, 0.f, 0.f, 0.f};
  for (int k0 = 0; k0 < Ec; k0 += 32)
    c = wmma_bf16(loadA_bf(pa, Ec, k0), loadB_f32(pb, Ec, k0), c);
  int lane = threadIdx.x & 31;
  int hi = lane >> 4, n = lane & 15;
#pragma unroll
  for (int r = 0; r < 8; ++r)
    out[(size_t)(m0 + hi * 8 + r) * Ec + n0 + n] = c[r];
}

extern "C" void kernel_launch(void* const* d_in, const int* in_sizes, int n_in,
                              void* d_out, int out_size, void* d_ws, size_t ws_size,
                              hipStream_t stream) {
  const float* query = (const float*)d_in[0];
  const float* key   = (const float*)d_in[1];
  const float* value = (const float*)d_in[2];
  const float* w_q   = (const float*)d_in[3];
  const float* w_k   = (const float*)d_in[4];
  const float* w_v   = (const float*)d_in[5];
  const float* w_o   = (const float*)d_in[6];

  float* out = (float*)d_out;                          // attn_out: B*T*E
  float* wts = out + (size_t)Bc * Tc * Ec;             // attn_weights: B*H*T*S

  const size_t headElems = (size_t)Bc * Hc * Tc * Dc;  // 4M bf16 elems each
  __bf16* qb  = (__bf16*)d_ws;
  __bf16* kb  = qb + headElems;
  __bf16* vt  = kb + headElems;
  __bf16* ctx = vt + headElems;                        // (B*T, E) bf16

  const int projGrid = ((Bc * Tc) / 16) * (Ec / 128);  // 2048
  qkv_proj_kernel<<<projGrid, 256, 0, stream>>>(query, w_q, qb, 0);
  qkv_proj_kernel<<<projGrid, 256, 0, stream>>>(key,   w_k, kb, 1);
  qkv_proj_kernel<<<projGrid, 256, 0, stream>>>(value, w_v, vt, 2);

  const int scoreGrid = (Bc * Hc) * (Tc / 16) * (Sc / 128);  // 65536
  scores_kernel<<<scoreGrid, 256, 0, stream>>>(qb, kb, wts);

  topk_softmax_kernel<<<Bc * Hc * Tc, 256, 0, stream>>>(wts);

  context_kernel<<<(Bc * Hc) * (Tc / 16), 128, 0, stream>>>(wts, vt, ctx);

  oproj_kernel<<<projGrid, 256, 0, stream>>>(ctx, w_o, out);
}
